// SimpleLayer_47974784697236
// MI455X (gfx1250) — compile-verified, matrix operand order checked by reading, not run on hardware
//
#include <hip/hip_runtime.h>
#include <stdint.h>

// ---------------- static problem config (matches reference) ----------------
#define BB 4
#define SS 2048
#define DD 2048
#define HH 16
#define DH 128
#define MAXLEN 1024
#define HDH (HH * DH)   // 2048

typedef __attribute__((ext_vector_type(16))) __bf16 v16bf;
typedef __attribute__((ext_vector_type(8)))  float  v8f;
typedef int v4i __attribute__((ext_vector_type(4)));

union Frag {                 // one 16x32 bf16 WMMA operand per lane (8 VGPRs)
  v16bf v;
  uint4 q[2];
  unsigned short u[16];
};

__device__ __forceinline__ unsigned short f2bf(float f) {
  unsigned int x = __builtin_bit_cast(unsigned int, f);
  unsigned int r = x + 0x7FFFu + ((x >> 16) & 1u);   // round-to-nearest-even
  return (unsigned short)(r >> 16);
}
__device__ __forceinline__ float bf2f(unsigned short h) {
  unsigned int x = ((unsigned int)h) << 16;
  return __builtin_bit_cast(float, x);
}
__device__ __forceinline__ v8f wmma_bf16(v16bf a, v16bf b, v8f c) {
  // D = A(16x32 bf16) * B(32x16 bf16) + C(16x16 f32)
  return __builtin_amdgcn_wmma_f32_16x16x32_bf16(false, a, false, b,
                                                 (short)0, c, false, false);
}

// -------- async global->LDS staging (gfx1250 GLOBAL_LOAD_ASYNC_TO_LDS_B128) --
typedef __attribute__((address_space(1))) v4i glob_v4i;  // global (AS1) int4
typedef __attribute__((address_space(3))) v4i lds_v4i;   // LDS (AS3) int4

#if defined(__has_builtin)
#if __has_builtin(__builtin_amdgcn_global_load_async_to_lds_b128)
#define HAVE_ASYNC_LDS 1
#endif
#endif

__device__ __forceinline__ void cp16(const unsigned short* g, unsigned short* l) {
#if defined(HAVE_ASYNC_LDS)
  __builtin_amdgcn_global_load_async_to_lds_b128((glob_v4i*)g, (lds_v4i*)l, 0, 0);
#else
  *(uint4*)l = *(const uint4*)g;
#endif
}

#if defined(HAVE_ASYNC_LDS)
#if __has_builtin(__builtin_amdgcn_s_wait_asynccnt)
#define WAIT_ASYNC(n) __builtin_amdgcn_s_wait_asynccnt(n)
#else
#define WAIT_ASYNC(n) asm volatile("s_wait_asynccnt %0" ::"i"(n) : "memory")
#endif
#else
#define WAIT_ASYNC(n) ((void)0)
#endif

// ======================================================================
// Elementwise f32 -> bf16 (4 elems / thread)
// ======================================================================
__global__ __launch_bounds__(256) void cvt_bf16(const float* __restrict__ x,
                                                unsigned short* __restrict__ y) {
  size_t i = (size_t)blockIdx.x * 256 + threadIdx.x;
  float4 d = *(const float4*)(x + i * 4);
  ushort4 p; p.x = f2bf(d.x); p.y = f2bf(d.y); p.z = f2bf(d.z); p.w = f2bf(d.w);
  *(ushort4*)(y + i * 4) = p;
}

// ======================================================================
// W[K][N] f32  ->  Wt[N][K] bf16   (32x32 tiles through LDS, both sides coalesced)
// grid (N/32, K/32), block 256 = 32x8
// ======================================================================
__global__ __launch_bounds__(256) void transpose_w(const float* __restrict__ W,
                                                   unsigned short* __restrict__ Wt,
                                                   int K, int N) {
  __shared__ float t[32][33];
  const int n0 = blockIdx.x * 32, k0 = blockIdx.y * 32;
  const int tx = threadIdx.x & 31, ty = threadIdx.x >> 5;
#pragma unroll
  for (int j = 0; j < 4; ++j)
    t[ty + 8 * j][tx] = W[(size_t)(k0 + ty + 8 * j) * N + n0 + tx];
  __syncthreads();
#pragma unroll
  for (int j = 0; j < 4; ++j)
    Wt[(size_t)(n0 + ty + 8 * j) * K + k0 + tx] = f2bf(t[tx][ty + 8 * j]);
}

// ======================================================================
// GEMM: C[M][N] = A[M][K] * Wt[N][K]^T, all-bf16 inputs, f32 accumulate.
// Tiles 128x128, k-step 32, double-buffered LDS fed by async global->LDS.
// 256 threads = 8 waves (4Mx2N); wave computes 32x64 via 2x4 WMMA accs.
// OUTMODE: 0 = f32 row-major, 1 = bf16 row-major, 2 = bf16 [b][h][dh][s]
// ======================================================================
template <int OUTMODE>
__global__ __launch_bounds__(256) void gemm_bf16(
    const unsigned short* __restrict__ A, const unsigned short* __restrict__ Bt,
    void* __restrict__ Cptr, int M, int K, int N) {
  __shared__ __align__(16) unsigned short As[2][128][40];  // [m][k], 80B pitch
  __shared__ __align__(16) unsigned short Bs[2][128][40];  // [n][k]

  const int tid  = threadIdx.x;
  const int lane = tid & 31, wv = tid >> 5;
  const int hl   = lane >> 4, lrow = lane & 15;
  const int wm0  = (wv >> 1) * 32, wn0 = (wv & 1) * 64;
  const int m0   = blockIdx.y * 128, n0 = blockIdx.x * 128;

  v8f acc[2][4];
#pragma unroll
  for (int i = 0; i < 2; ++i)
#pragma unroll
    for (int j = 0; j < 4; ++j) acc[i][j] = (v8f){0.f,0.f,0.f,0.f,0.f,0.f,0.f,0.f};

  // 4 chunk copies per thread per tile pair (2 A + 2 B), 16B each
  auto issue = [&](int buf, int k0) {
#pragma unroll
    for (int i = 0; i < 2; ++i) {
      int idx = tid + 256 * i;                  // 512 chunks of 8 ushorts
      int r = idx >> 2, c8 = (idx & 3) * 8;
      cp16(A + (size_t)(m0 + r) * K + k0 + c8, &As[buf][r][c8]);
    }
#pragma unroll
    for (int i = 0; i < 2; ++i) {
      int idx = tid + 256 * i;
      int r = idx >> 2, c8 = (idx & 3) * 8;
      cp16(Bt + (size_t)(n0 + r) * K + k0 + c8, &Bs[buf][r][c8]);
    }
  };

  const int nsteps = K >> 5;
  issue(0, 0);
  for (int s = 0; s < nsteps; ++s) {
    const int buf = s & 1;
    if (s + 1 < nsteps) { issue(buf ^ 1, (s + 1) * 32); WAIT_ASYNC(4); }
    else                { WAIT_ASYNC(0); }
    __syncthreads();

    Frag a[2];
#pragma unroll
    for (int mt = 0; mt < 2; ++mt) {
      const unsigned short* base = &As[buf][wm0 + mt * 16 + lrow][0];
      a[mt].q[0] = *(const uint4*)(base + 8 * hl);
      a[mt].q[1] = *(const uint4*)(base + 16 + 8 * hl);
    }
#pragma unroll
    for (int nt = 0; nt < 4; ++nt) {
      Frag bfr;
      const unsigned short* base = &Bs[buf][wn0 + nt * 16 + lrow][0];
      bfr.q[0] = *(const uint4*)(base + 8 * hl);
      bfr.q[1] = *(const uint4*)(base + 16 + 8 * hl);
#pragma unroll
      for (int mt = 0; mt < 2; ++mt)
        acc[mt][nt] = wmma_bf16(a[mt].v, bfr.v, acc[mt][nt]);
    }
    __syncthreads();
  }

  // C-frag layout: VGPR r / lane -> row r+8*hl, col lrow
#pragma unroll
  for (int mt = 0; mt < 2; ++mt)
#pragma unroll
    for (int nt = 0; nt < 4; ++nt) {
      int col = n0 + wn0 + nt * 16 + lrow;
#pragma unroll
      for (int r = 0; r < 8; ++r) {
        int row = m0 + wm0 + mt * 16 + r + 8 * hl;
        float vv = acc[mt][nt][r];
        if (OUTMODE == 0) {
          ((float*)Cptr)[(size_t)row * N + col] = vv;
        } else if (OUTMODE == 1) {
          ((unsigned short*)Cptr)[(size_t)row * N + col] = f2bf(vv);
        } else {  // V output -> [b][h][dh][s] for transpose-free attention staging
          int bb = row >> 11, si = row & (SS - 1);   // M = b*SS + s
          int hh = col >> 7,  dh = col & (DH - 1);   // N = h*DH + dh
          ((unsigned short*)Cptr)[(((size_t)(bb * HH + hh)) * DH + dh) * SS + si] =
              f2bf(vv);
        }
      }
    }
}

// ======================================================================
// Ragged gather: order[b][l] = l-th kept position (stable), -1 past n_b.
// ======================================================================
__global__ void build_order(const unsigned char* __restrict__ skip_mask,
                            int* __restrict__ order, int* __restrict__ nvals) {
  int b = threadIdx.x;
  if (b >= BB) return;
  const unsigned char* m = skip_mask + (size_t)b * SS;
  int cnt = 0;
  for (int s = 0; s < SS && cnt < MAXLEN; ++s)
    if (m[s]) order[b * MAXLEN + cnt++] = s;
  nvals[b] = cnt;
  for (int l = cnt; l < MAXLEN; ++l) order[b * MAXLEN + l] = -1;
}

// ======================================================================
// RoPE in place on bf16 [BB][L][HH][DH]; positions = gathered (q) or full (k).
// ======================================================================
__global__ __launch_bounds__(256) void rope_kernel(
    unsigned short* __restrict__ x, const int* __restrict__ order,
    int L, int useOrder) {
  size_t idx = (size_t)blockIdx.x * 256 + threadIdx.x;  // over B*L*H*64
  int i = (int)(idx & 63);
  size_t row = idx >> 6;                // b*L*H + l*H + h
  size_t bl = row / HH;
  int l = (int)(bl % L), b = (int)(bl / L);
  int pos;
  if (useOrder) { int o = order[b * MAXLEN + l]; pos = (o < 0) ? 0 : o; }
  else pos = l;
  float inv = __powf(10000.0f, -(float)i * (1.0f / 64.0f));
  float ang = (float)pos * inv;
  float s, c; __sincosf(ang, &s, &c);
  unsigned short* p = x + row * DH;
  float x1 = bf2f(p[i]), x2 = bf2f(p[i + 64]);
  p[i]      = f2bf(x1 * c - x2 * s);
  p[i + 64] = f2bf(x2 * c + x1 * s);
}

// ======================================================================
// Flash attention. Grid (qblk=8, h=16, b=4), 256 threads = 8 waves.
// Wave owns 16 query rows x DH=128. Double-buffered async K/V tiles (KV-step
// 32) shared by the workgroup. V comes pre-transposed ([b][h][dh][s]).
// Causal mask on gathered original positions; padded rows -> zero.
// ======================================================================
__global__ __launch_bounds__(256) void attn_kernel(
    const unsigned short* __restrict__ qh, const unsigned short* __restrict__ kh,
    const unsigned short* __restrict__ vt, unsigned short* __restrict__ ctx,
    const int* __restrict__ order, const int* __restrict__ nvals) {
  __shared__ __align__(16) unsigned short Ks[2][32][136];   // [kv][dh]
  __shared__ __align__(16) unsigned short Vs[2][128][40];   // [dh][kv]
  __shared__ __align__(16) unsigned short Ps[8][16][40];    // per-wave P staging
  __shared__ int red[256];

  const int tid  = threadIdx.x;
  const int lane = tid & 31, wv = tid >> 5;
  const int hl   = lane >> 4, lrow = lane & 15;
  const int q0 = blockIdx.x * 128;
  const int h  = blockIdx.y, b = blockIdx.z;
  const int n_b = nvals[b];

  // ---- block-uniform KV trip count: max gathered position in block ----
  int ov = -1;
  if (tid < 128) { int l = q0 + tid; ov = (l < n_b) ? order[b * MAXLEN + l] : -1; }
  red[tid] = ov;
  __syncthreads();
  for (int off = 128; off >= 1; off >>= 1) {
    if (tid < off) { int x0 = red[tid], x1 = red[tid + off]; red[tid] = x0 > x1 ? x0 : x1; }
    __syncthreads();
  }
  const int kv_limit = ((red[0] + 1) + 31) & ~31;   // 0 if nothing valid
  __syncthreads();

  // ---- per-lane query metadata for the 8 C-frag rows ----
  int  qord[8];
  bool qval[8];
#pragma unroll
  for (int r = 0; r < 8; ++r) {
    int l = q0 + wv * 16 + r + 8 * hl;
    int o = (l < n_b) ? order[b * MAXLEN + l] : -1;
    qord[r] = o;
    qval[r] = (o >= 0);
  }

  // ---- preload Q A-frags (16x128 per wave -> 4 frags of 16x32) ----
  Frag qa[4];
  const size_t qbase =
      ((size_t)b * MAXLEN + (q0 + wv * 16 + lrow)) * HDH + (size_t)h * DH;
#pragma unroll
  for (int d = 0; d < 4; ++d) {
    qa[d].q[0] = *(const uint4*)(qh + qbase + d * 32 + 8 * hl);
    qa[d].q[1] = *(const uint4*)(qh + qbase + d * 32 + 16 + 8 * hl);
  }

  float rm[8], rl[8];
  v8f accd[8];
#pragma unroll
  for (int r = 0; r < 8; ++r) { rm[r] = -1e30f; rl[r] = 0.f; }
#pragma unroll
  for (int d = 0; d < 8; ++d) accd[d] = (v8f){0.f,0.f,0.f,0.f,0.f,0.f,0.f,0.f};

  const unsigned short* Kb = kh + (size_t)b * SS * HDH + (size_t)h * DH;
  const unsigned short* Vb = vt + ((size_t)(b * HH + h)) * DH * SS;   // [dh][s]
  const float scale = 0.08838834764831845f;  // 1/sqrt(128)

  // 4 async chunk copies per thread per KV tile (2 K + 2 V)
  auto issueKV = [&](int buf, int kv0) {
#pragma unroll
    for (int i = 0; i < 2; ++i) {
      int idx = tid + 256 * i;                  // K: 32x128 = 512 chunks
      int r = idx >> 4, c8 = (idx & 15) * 8;
      cp16(Kb + (size_t)(kv0 + r) * HDH + c8, &Ks[buf][r][c8]);
    }
#pragma unroll
    for (int i = 0; i < 2; ++i) {
      int idx = tid + 256 * i;                  // V: 128x32 = 512 chunks
      int r = idx >> 2, c8 = (idx & 3) * 8;
      cp16(Vb + (size_t)r * SS + kv0 + c8, &Vs[buf][r][c8]);
    }
  };

  const int nst = kv_limit >> 5;
  if (nst > 0) issueKV(0, 0);
  for (int st = 0; st < nst; ++st) {
    const int buf = st & 1;
    const int kv0 = st * 32;
    if (st + 1 < nst) { issueKV(buf ^ 1, (st + 1) * 32); WAIT_ASYNC(4); }
    else              { WAIT_ASYNC(0); }
    __syncthreads();

    // ---- logits: Q(16x128) * K^T(128x32) = 8 WMMAs ----
    v8f lf[2];
    lf[0] = (v8f){0.f,0.f,0.f,0.f,0.f,0.f,0.f,0.f};
    lf[1] = (v8f){0.f,0.f,0.f,0.f,0.f,0.f,0.f,0.f};
#pragma unroll
    for (int nt = 0; nt < 2; ++nt)
#pragma unroll
      for (int d = 0; d < 4; ++d) {
        Frag kb;
        const unsigned short* base = &Ks[buf][nt * 16 + lrow][d * 32];
        kb.q[0] = *(const uint4*)(base + 8 * hl);
        kb.q[1] = *(const uint4*)(base + 16 + 8 * hl);
        lf[nt] = wmma_bf16(qa[d].v, kb.v, lf[nt]);
      }

    // ---- online softmax per row (16-lane shfl reductions) ----
    const int s0 = kv0 + lrow, s1 = kv0 + 16 + lrow;
#pragma unroll
    for (int r = 0; r < 8; ++r) {
      float v0 = lf[0][r] * scale, v1 = lf[1][r] * scale;
      if (!qval[r] || s0 > qord[r]) v0 = -1e30f;
      if (!qval[r] || s1 > qord[r]) v1 = -1e30f;
      float mx = fmaxf(v0, v1);
#pragma unroll
      for (int off = 1; off < 16; off <<= 1) mx = fmaxf(mx, __shfl_xor(mx, off, 32));
      float newm = fmaxf(rm[r], mx);
      float alpha = __expf(rm[r] - newm);
      float p0 = (v0 <= -1e29f) ? 0.f : __expf(v0 - newm);
      float p1 = (v1 <= -1e29f) ? 0.f : __expf(v1 - newm);
      float ps = p0 + p1;
#pragma unroll
      for (int off = 1; off < 16; off <<= 1) ps += __shfl_xor(ps, off, 32);
      rl[r] = rl[r] * alpha + ps;
      rm[r] = newm;
#pragma unroll
      for (int d = 0; d < 8; ++d) accd[d][r] *= alpha;
      // stage P (C-layout -> A-layout conversion via per-wave LDS)
      Ps[wv][r + 8 * hl][lrow]      = f2bf(p0);
      Ps[wv][r + 8 * hl][16 + lrow] = f2bf(p1);
    }
    asm volatile("s_wait_dscnt 0" ::: "memory");  // wave-local LDS RAW fence

    // ---- ctx += P(16x32) * V(32x128) = 8 WMMAs ----
    Frag pa;
    {
      const unsigned short* base = &Ps[wv][lrow][0];
      pa.q[0] = *(const uint4*)(base + 8 * hl);
      pa.q[1] = *(const uint4*)(base + 16 + 8 * hl);
    }
#pragma unroll
    for (int d = 0; d < 8; ++d) {
      Frag vb;
      const unsigned short* base = &Vs[buf][d * 16 + lrow][0];
      vb.q[0] = *(const uint4*)(base + 8 * hl);
      vb.q[1] = *(const uint4*)(base + 16 + 8 * hl);
      accd[d] = wmma_bf16(pa.v, vb.v, accd[d]);
    }
    __syncthreads();
  }

  // ---- normalize + store ctx (zero for padded rows) ----
#pragma unroll
  for (int d = 0; d < 8; ++d)
#pragma unroll
    for (int r = 0; r < 8; ++r) {
      int l = q0 + wv * 16 + r + 8 * hl;
      float o = (qval[r] && rl[r] > 0.f) ? accd[d][r] / rl[r] : 0.f;
      ctx[((size_t)b * MAXLEN + l) * HDH + (size_t)h * DH + d * 16 + lrow] = f2bf(o);
    }
}

// ======================================================================
extern "C" void kernel_launch(void* const* d_in, const int* in_sizes, int n_in,
                              void* d_out, int out_size, void* d_ws, size_t ws_size,
                              hipStream_t stream) {
  const float* q_src = (const float*)d_in[0];
  const float* k_src = (const float*)d_in[1];
  const float* v_src = (const float*)d_in[2];
  const float* Wq    = (const float*)d_in[3];
  const float* Wk    = (const float*)d_in[4];
  const float* Wv    = (const float*)d_in[5];
  const float* Wo    = (const float*)d_in[6];
  // d_in[7] = mask0 (causal tril) — reconstructed analytically, not read.
  // d_in[8] = pos_full (arange)   — reconstructed analytically, not read.
  const unsigned char* skip = (const unsigned char*)d_in[9];  // numpy bool8
  float* out = (float*)d_out;

  const int Mq = BB * MAXLEN;  // 4096
  const int Mk = BB * SS;      // 8192

  // workspace carve-up (all 16B-aligned segments); ctx aliases qsb (dead then)
  char* ws = (char*)d_ws;
  unsigned short* qsb = (unsigned short*)ws; ws += (size_t)Mq * DD * 2;   // 16.8 MB
  unsigned short* ksb = (unsigned short*)ws; ws += (size_t)Mk * DD * 2;   // 33.6 MB
  unsigned short* vsb = (unsigned short*)ws; ws += (size_t)Mk * DD * 2;   // 33.6 MB
  unsigned short* Wqt = (unsigned short*)ws; ws += (size_t)DD * HDH * 2;  // 8.4 MB
  unsigned short* Wkt = (unsigned short*)ws; ws += (size_t)DD * HDH * 2;
  unsigned short* Wvt = (unsigned short*)ws; ws += (size_t)DD * HDH * 2;
  unsigned short* Wot = (unsigned short*)ws; ws += (size_t)HDH * DD * 2;
  unsigned short* qh  = (unsigned short*)ws; ws += (size_t)Mq * HDH * 2;  // 16.8 MB
  unsigned short* kh  = (unsigned short*)ws; ws += (size_t)Mk * HDH * 2;  // 33.6 MB
  unsigned short* vtb = (unsigned short*)ws; ws += (size_t)Mk * HDH * 2;  // 33.6 MB
  int* order = (int*)ws; ws += (size_t)BB * MAXLEN * 4;
  int* nvals = (int*)ws;
  unsigned short* ctx = qsb;   // alias: qsb dead after Q projection

  // ---- one-time bf16 conversion + weight transposes ----
  cvt_bf16<<<(Mq * DD / 4) / 256, 256, 0, stream>>>(q_src, qsb);
  cvt_bf16<<<(Mk * DD / 4) / 256, 256, 0, stream>>>(k_src, ksb);
  cvt_bf16<<<(Mk * DD / 4) / 256, 256, 0, stream>>>(v_src, vsb);
  transpose_w<<<dim3(HDH / 32, DD / 32), 256, 0, stream>>>(Wq, Wqt, DD, HDH);
  transpose_w<<<dim3(HDH / 32, DD / 32), 256, 0, stream>>>(Wk, Wkt, DD, HDH);
  transpose_w<<<dim3(HDH / 32, DD / 32), 256, 0, stream>>>(Wv, Wvt, DD, HDH);
  transpose_w<<<dim3(DD / 32, HDH / 32), 256, 0, stream>>>(Wo, Wot, HDH, DD);
  build_order<<<1, 32, 0, stream>>>(skip, order, nvals);

  // ---- projections (WMMA bf16) ----
  gemm_bf16<1><<<dim3(HDH / 128, Mq / 128), 256, 0, stream>>>(qsb, Wqt, qh, Mq, DD, HDH);
  gemm_bf16<1><<<dim3(HDH / 128, Mk / 128), 256, 0, stream>>>(ksb, Wkt, kh, Mk, DD, HDH);
  gemm_bf16<2><<<dim3(HDH / 128, Mk / 128), 256, 0, stream>>>(vsb, Wvt, vtb, Mk, DD, HDH);

  // ---- RoPE ----
  rope_kernel<<<(BB * MAXLEN * HH * 64) / 256, 256, 0, stream>>>(qh, order, MAXLEN, 1);
  rope_kernel<<<(BB * SS * HH * 64) / 256, 256, 0, stream>>>(kh, order, SS, 0);

  // ---- flash attention ----
  attn_kernel<<<dim3(MAXLEN / 128, HH, BB), 256, 0, stream>>>(
      qh, kh, vtb, ctx, order, nvals);

  // ---- output projection -> f32 ----
  gemm_bf16<0><<<dim3(DD / 128, Mq / 128), 256, 0, stream>>>(ctx, Wot, out, Mq, HDH, DD);
}
